// CoreAttention_83099027243601
// MI455X (gfx1250) — matrix operand order repaired and checked by reading, not compiled, not run on hardware
//
#include <hip/hip_runtime.h>

typedef _Float16 v16h __attribute__((ext_vector_type(16)));
typedef _Float16 v8h  __attribute__((ext_vector_type(8)));
typedef _Float16 v4h  __attribute__((ext_vector_type(4)));
typedef float    v8f  __attribute__((ext_vector_type(8)));

#define SQ_ 2048
#define SK_ 2048
#define B_  2
#define H_  16
#define D_  64
#define NT64_ (SK_ / 64)                  // 32 key-tiles (64 keys) per head
// softmax in base-2 domain; scale (1/8)*log2(e) is folded into the Q fragments
#define SCALE2_ 0.18033688011112042f
#define MASK2_  (-14426.950408889634f)    // -10000 * log2(e)

// ---------------- wave32 xor-reductions via v_permlane16_b32 (VALU, no LDS) ----
#define XM1_LO 0x67452301u
#define XM1_HI 0xEFCDAB89u
#define XM2_LO 0x54761032u
#define XM2_HI 0xDCFE98BAu
#define XM4_LO 0x32107654u
#define XM4_HI 0xBA98FEDCu
#define XM8_LO 0xFEDCBA98u
#define XM8_HI 0x76543210u

__device__ __forceinline__ float permx16(float x, unsigned lo, unsigned hi) {
    unsigned xi = (unsigned)__float_as_int(x);
    unsigned y  = __builtin_amdgcn_permlane16(xi, xi, lo, hi, false, false);
    return __int_as_float((int)y);
}
__device__ __forceinline__ float rowmax16(float x) {
    x = fmaxf(x, permx16(x, XM1_LO, XM1_HI));
    x = fmaxf(x, permx16(x, XM2_LO, XM2_HI));
    x = fmaxf(x, permx16(x, XM4_LO, XM4_HI));
    x = fmaxf(x, permx16(x, XM8_LO, XM8_HI));
    return x;
}
__device__ __forceinline__ float rowsum16(float x) {
    x += permx16(x, XM1_LO, XM1_HI);
    x += permx16(x, XM2_LO, XM2_HI);
    x += permx16(x, XM4_LO, XM4_HI);
    x += permx16(x, XM8_LO, XM8_HI);
    return x;
}

// ---------------- async global -> LDS copy (CDNA5, tracked on ASYNCcnt) -------
__device__ __forceinline__ void async_cp_b128(unsigned lds_off, const void* sbase,
                                              unsigned goff) {
    asm volatile("global_load_async_to_lds_b128 %0, %1, %2"
                 :: "v"(lds_off), "v"(goff), "s"(sbase) : "memory");
}
__device__ __forceinline__ void wait_asynccnt0() {
    asm volatile("s_wait_asynccnt 0x0" ::: "memory");
}

// Contracted-dim permutation for the PV product: position p <-> key 16*(p&3)+(p>>2).
// P is stored so each lane's 4 probs (keys n, n+16, n+32, n+48) are 4 adjacent halfs;
// the V tile's k-order uses the same permutation, so the WMMA contraction matches.

// =============================================================================
// Pre-pass: convert K/V (f32) once into f16 64-key tiles pre-swizzled into the
// WMMA B-fragment layout this code uses (contiguous k), 8KB per tile, contiguous in d_ws.
//   B layout used here (contiguous k, not AMD's table order): element (k, n) -> lane = n + 16*(k>>4), half = k&15
//   K tile: fragments (s = key block 0..3, c = d chunk 0..1): ((s*2+c)*32+L)*16+h
//   V tile: fragments (nc = d chunk 0..3, kc = k chunk 0..1): ((nc*2+kc)*32+L)*16+h
//           with k-index = permuted position (see above)
// =============================================================================
__global__ __launch_bounds__(128)
void preswizzle_kv(const float* __restrict__ K, const float* __restrict__ V,
                   _Float16* __restrict__ Kh, _Float16* __restrict__ Vh)
{
    __shared__ _Float16 Vt[64 * 64] __attribute__((aligned(32)));   // 8 KB

    const int it = blockIdx.x;            // 64-key tile
    const int bh = blockIdx.y;            // head
    const int t  = threadIdx.x;
    const int kb = it * 64;
    const size_t rowstride = (size_t)B_ * H_ * D_;
    const size_t headoff   = (size_t)bh * D_;
    const size_t tilebase  = (size_t)(bh * NT64_ + it) * 4096;      // halfs

    const int r  = t >> 1;                // key row 0..63
    const int cs = (t & 1) * 32;          // col start 0/32

    // ---- K: coalesced read -> two contiguous 32B swizzled writes ----
    {
        const float* krow = K + (size_t)(kb + r) * rowstride + headoff + cs;
        const int s  = r >> 4;
        const int nk = r & 15;
        #pragma unroll
        for (int c16 = 0; c16 < 2; ++c16) {
            const int d0 = cs + c16 * 16;
            const int c  = d0 >> 5;
            const int gc = (d0 >> 4) & 1;
            v16h pk;
            #pragma unroll
            for (int i = 0; i < 16; ++i) pk[i] = (_Float16)krow[c16 * 16 + i];
            *(v16h*)&Kh[tilebase + (((s * 2 + c) * 32) + nk + 16 * gc) * 16] = pk;
        }
    }
    // ---- V: coalesced read -> LDS bounce -> contiguous swizzled writes ----
    {
        const float* vrow = V + (size_t)(kb + r) * rowstride + headoff + cs;
        #pragma unroll
        for (int i = 0; i < 32; ++i) Vt[r * 64 + cs + i] = (_Float16)vrow[i];
    }
    __syncthreads();
    #pragma unroll
    for (int rep = 0; rep < 2; ++rep) {   // 256 16-half chunks / 128 threads
        const int ch   = t + rep * 128;
        const int frag = ch >> 5;         // nc*2 + kc
        const int nc   = frag >> 1;
        const int kc   = frag & 1;
        const int L    = ch & 31;
        const int n    = L & 15;
        const int gk   = L >> 4;
        v16h pv;
        #pragma unroll
        for (int h = 0; h < 16; ++h) {
            const int p   = kc * 32 + 16 * gk + h;        // permuted position
            const int key = 16 * (p & 3) + (p >> 2);
            pv[h] = Vt[key * 64 + nc * 16 + n];
        }
        *(v16h*)&Vh[tilebase + (frag * 32 + L) * 16] = pv;
    }
}

// =============================================================================
// Flash-attention forward, causal. 4 wave32 per block; wave w owns Q rows
// [qblock+16w, +16); 64-key tiles. PRE: double-buffered async-DMA staging of
// pre-swizzled f16 tiles (copy of tile i+1 overlaps compute of tile i).
// Masking only on the final (diagonal) tile — block-uniform.
// =============================================================================
template<bool PRE>
__global__ __launch_bounds__(128)
void fa_fwd_causal_wmma(const float* __restrict__ Q,
                        const float* __restrict__ Kf, const float* __restrict__ Vf,
                        const _Float16* __restrict__ Kh, const _Float16* __restrict__ Vh,
                        float* __restrict__ O)
{
    __shared__ _Float16 Kbuf[2][4096] __attribute__((aligned(32)));   // 16 KB
    __shared__ _Float16 Vbuf[2][4096] __attribute__((aligned(32)));   // 16 KB
    __shared__ _Float16 Plds[4 * 1024] __attribute__((aligned(32)));  //  8 KB

    const int qblock = blockIdx.x * 64;
    const int bh     = blockIdx.y;

    const int t    = threadIdx.x;
    const int wv   = t >> 5;
    const int lane = t & 31;
    const int n16  = lane & 15;
    const int g    = lane >> 4;

    const int    qtile     = qblock + wv * 16;
    const size_t rowstride = (size_t)B_ * H_ * D_;
    const size_t headoff   = (size_t)bh * D_;

    // ---- Q tile as two A-fragments, softmax scale folded in ----
    v16h aq[2];
    {
        const float* qrow = Q + (size_t)(qtile + n16) * rowstride + headoff;
        #pragma unroll
        for (int c = 0; c < 2; ++c) {
            const int base = c * 32 + 16 * g;
            #pragma unroll
            for (int i = 0; i < 8; ++i) {
                aq[c][i]     = (_Float16)(qrow[base + i]      * SCALE2_);
                aq[c][i + 8] = (_Float16)(qrow[base + 8 + i] * SCALE2_);
            }
        }
    }

    v8f acc[4];
    #pragma unroll
    for (int nc = 0; nc < 4; ++nc) acc[nc] = (v8f){};
    float mrow[8], lrow[8];               // lrow: per-lane PARTIAL sums
    #pragma unroll
    for (int v = 0; v < 8; ++v) { mrow[v] = -1.0e30f; lrow[v] = 0.0f; }

    const int ntiles = qblock / 64 + 1;   // causal bound; last tile is diagonal

    auto issue_tile = [&](int it2, int sel) {
        const char* kt = (const char*)Kh + (((size_t)(bh * NT64_ + it2)) << 13);
        const char* vt = (const char*)Vh + (((size_t)(bh * NT64_ + it2)) << 13);
        const unsigned ko = (unsigned)(uintptr_t)&Kbuf[sel][0];
        const unsigned vo = (unsigned)(uintptr_t)&Vbuf[sel][0];
        #pragma unroll
        for (int j = 0; j < 4; ++j) {
            const unsigned o = t * 16 + j * 2048;
            async_cp_b128(ko + o, kt, o);
            async_cp_b128(vo + o, vt, o);
        }
    };

    if constexpr (PRE) { issue_tile(0, 0); }

    for (int it = 0; it < ntiles; ++it) {
        const int kb = it * 64;
        int cur;
        if constexpr (PRE) {
            wait_asynccnt0();             // my copies for tile `it` landed
            __syncthreads();              // everyone's landed; prior reads done
            cur = it & 1;
            if (it + 1 < ntiles) issue_tile(it + 1, (it + 1) & 1);
        } else {
            __syncthreads();
            const int r  = t >> 1;        // key row 0..63
            const int cs = (t & 1) * 32;
            {   // K stage
                const float* krow = Kf + (size_t)(kb + r) * rowstride + headoff + cs;
                const int s  = r >> 4;
                const int nk = r & 15;
                #pragma unroll
                for (int i = 0; i < 32; ++i) {
                    const int d = cs + i;
                    Kbuf[0][((s * 2 + (d >> 5)) * 32 + nk + 16 * ((d >> 4) & 1)) * 16 + (d & 15)]
                        = (_Float16)krow[i];
                }
            }
            {   // V stage (permuted k-order)
                const float* vrow = Vf + (size_t)(kb + r) * rowstride + headoff + cs;
                const int p    = ((r & 15) * 4) | (r >> 4);
                const int kc   = p >> 5;
                const int kidx = p & 31;
                const int gk   = kidx >> 4;
                const int hk   = kidx & 15;
                #pragma unroll
                for (int i = 0; i < 32; ++i) {
                    const int d = cs + i;
                    Vbuf[0][(((d >> 4) * 2 + kc) * 32 + (d & 15) + 16 * gk) * 16 + hk]
                        = (_Float16)vrow[i];
                }
            }
            __syncthreads();
            cur = 0;
        }

        // ---- S = Q * K^T over 64 keys (4 C fragments) ----
        v8f sf[4];
        #pragma unroll
        for (int si = 0; si < 4; ++si) {
            const v16h b0 = *(const v16h*)&Kbuf[cur][((si * 2 + 0) * 32 + lane) * 16];
            const v16h b1 = *(const v16h*)&Kbuf[cur][((si * 2 + 1) * 32 + lane) * 16];
            v8f z = (v8f){};
            z = __builtin_amdgcn_wmma_f32_16x16x32_f16(false, aq[0], false, b0, (short)0, z, false, false);
            z = __builtin_amdgcn_wmma_f32_16x16x32_f16(false, aq[1], false, b1, (short)0, z, false, false);
            sf[si] = z;
        }

        const bool lastTile = (it == ntiles - 1);   // only tile needing the causal mask
        const int  pbase    = wv * 1024;
        #pragma unroll
        for (int v = 0; v < 8; ++v) {
            float s0 = sf[0][v], s1 = sf[1][v], s2 = sf[2][v], s3 = sf[3][v];
            if (lastTile) {
                const int row = qtile + v + 8 * g;   // C layout: M = v + 8g
                if (kb + n16 > row)      s0 = MASK2_;
                if (kb + 16 + n16 > row) s1 = MASK2_;
                if (kb + 32 + n16 > row) s2 = MASK2_;
                if (kb + 48 + n16 > row) s3 = MASK2_;
            }
            const float mx   = rowmax16(fmaxf(fmaxf(s0, s1), fmaxf(s2, s3)));
            const float mnew = fmaxf(mrow[v], mx);
            const float corr = __builtin_amdgcn_exp2f(mrow[v] - mnew);
            const float p0 = __builtin_amdgcn_exp2f(s0 - mnew);
            const float p1 = __builtin_amdgcn_exp2f(s1 - mnew);
            const float p2 = __builtin_amdgcn_exp2f(s2 - mnew);
            const float p3 = __builtin_amdgcn_exp2f(s3 - mnew);
            lrow[v] = lrow[v] * corr + ((p0 + p1) + (p2 + p3));
            mrow[v] = mnew;
            acc[0][v] *= corr; acc[1][v] *= corr;
            acc[2][v] *= corr; acc[3][v] *= corr;
            v4h pk;                                   // 4 adjacent halfs (permuted cols)
            pk[0] = (_Float16)p0; pk[1] = (_Float16)p1;
            pk[2] = (_Float16)p2; pk[3] = (_Float16)p3;
            *(v4h*)&Plds[pbase + (v + 8 * g) * 64 + n16 * 4] = pk;
        }

        // ---- reload P as two A-fragments (wave-private LDS, in-order per wave) ----
        v16h ap[2];
        #pragma unroll
        for (int kc = 0; kc < 2; ++kc) {
            const v8h lo = *(const v8h*)&Plds[pbase + n16 * 64 + kc * 32 + 16 * g];
            const v8h hi = *(const v8h*)&Plds[pbase + n16 * 64 + kc * 32 + 16 * g + 8];
            #pragma unroll
            for (int i = 0; i < 8; ++i) { ap[kc][i] = lo[i]; ap[kc][i + 8] = hi[i]; }
        }

        // ---- O += P * V ----
        #pragma unroll
        for (int nc = 0; nc < 4; ++nc) {
            const v16h bv0 = *(const v16h*)&Vbuf[cur][((nc * 2 + 0) * 32 + lane) * 16];
            const v16h bv1 = *(const v16h*)&Vbuf[cur][((nc * 2 + 1) * 32 + lane) * 16];
            acc[nc] = __builtin_amdgcn_wmma_f32_16x16x32_f16(false, ap[0], false, bv0, (short)0, acc[nc], false, false);
            acc[nc] = __builtin_amdgcn_wmma_f32_16x16x32_f16(false, ap[1], false, bv1, (short)0, acc[nc], false, false);
        }
    }

    // ---- epilogue: finish row sums, normalize, store ----
    #pragma unroll
    for (int v = 0; v < 8; ++v) {
        const float inv = 1.0f / rowsum16(lrow[v]);
        const int   row = qtile + v + 8 * g;
        float* orow = O + (size_t)row * rowstride + headoff;
        #pragma unroll
        for (int nc = 0; nc < 4; ++nc)
            orow[nc * 16 + n16] = acc[nc][v] * inv;
    }
}

extern "C" void kernel_launch(void* const* d_in, const int* in_sizes, int n_in,
                              void* d_out, int out_size, void* d_ws, size_t ws_size,
                              hipStream_t stream) {
    (void)in_sizes; (void)n_in; (void)out_size;
    const float* Q = (const float*)d_in[0];
    const float* K = (const float*)d_in[1];
    const float* V = (const float*)d_in[2];
    // d_in[3] = attention_mask (bool) — ignored for causal mask type
    float* O = (float*)d_out;

    dim3 grid(SQ_ / 64, B_ * H_);
    dim3 block(128);

    const size_t halfs_per_tensor = (size_t)B_ * H_ * NT64_ * 4096;   // 8 MB
    if (ws_size >= 2 * halfs_per_tensor * sizeof(_Float16)) {
        _Float16* Kh = (_Float16*)d_ws;
        _Float16* Vh = Kh + halfs_per_tensor;
        preswizzle_kv<<<dim3(NT64_, B_ * H_), block, 0, stream>>>(K, V, Kh, Vh);
        fa_fwd_causal_wmma<true><<<grid, block, 0, stream>>>(Q, K, V, Kh, Vh, O);
    } else {
        fa_fwd_causal_wmma<false><<<grid, block, 0, stream>>>(Q, K, V, nullptr, nullptr, O);
    }
}
